// PointPlaneResnet_63788854280425
// MI455X (gfx1250) — matrix operand order repaired and checked by reading, not compiled, not run on hardware
//
#include <hip/hip_runtime.h>
#include <cstdint>
#include <cstddef>

// ---------------------------------------------------------------------------
// Types for CDNA5 WMMA (wave32, v_wmma_f32_16x16x32_bf16)
// ---------------------------------------------------------------------------
typedef __attribute__((ext_vector_type(16))) __bf16         v16bf;
typedef __attribute__((ext_vector_type(8)))  float          v8f;
typedef __attribute__((ext_vector_type(8)))  unsigned short v8u;
typedef __attribute__((ext_vector_type(16))) unsigned short v16u;

#define BATCH  2
#define NPTS   8192
#define ROWS   (BATCH * NPTS)
#define KNN_K  40
#define KP1    (KNN_K + 1)
#define BN_EPS 1e-5f

static __device__ __forceinline__ unsigned short f2bf(float f) {
  unsigned x = __float_as_uint(f);
  unsigned r = ((x >> 16) & 1u) + 0x7fffu;   // round-to-nearest-even
  return (unsigned short)((x + r) >> 16);
}
static __device__ __forceinline__ float bf2f(unsigned short u) {
  return __uint_as_float((unsigned)u << 16);
}

union FragU { v16u u; v16bf b; };

// A-matrix 16x32 bf16 per-lane layout: row M = lane%16,
// elements e -> K = k0 + 16*half + e
template <bool RELU>
static __device__ __forceinline__ v16bf load_fragA(const unsigned short* p) {
  v8u c0 = *(const v8u*)(p);
  v8u c1 = *(const v8u*)(p + 8);
  FragU cv;
#pragma unroll
  for (int e = 0; e < 8; ++e) { cv.u[e] = c0[e]; cv.u[e + 8] = c1[e]; }
  if (RELU) {
#pragma unroll
    for (int e = 0; e < 16; ++e)
      cv.u[e] = (cv.u[e] & 0x8000u) ? (unsigned short)0 : cv.u[e];
  }
  return cv.b;
}
// B-matrix 32x16 bf16: col N = lane%16, elements e -> K = k0 + 16*half + e
static __device__ __forceinline__ v16bf load_fragB(const unsigned short* p) {
  FragU cv; cv.u = *(const v16u*)p; return cv.b;
}

// Fragment set for one 32(M) x 64(N) x 32(K) step.
struct FragSet {
  v16bf a0, a1, b0, b1, b2, b3;
};

// ---------------------------------------------------------------------------
// f32 -> bf16 conversion
// ---------------------------------------------------------------------------
__global__ void cvt_f32_bf16(const float* __restrict__ in,
                             unsigned short* __restrict__ out, int n) {
  int i = blockIdx.x * blockDim.x + threadIdx.x;
  if (i < n) out[i] = f2bf(in[i]);
}

// ---------------------------------------------------------------------------
// KNN + plane-kernel feature.  One thread per query point; candidates tiled
// through LDS; top-(k+1) kept via replace-max; second pass accumulates
// neighbor coordinate sums (feature is linear in neighbor coords).
// ---------------------------------------------------------------------------
__global__ __launch_bounds__(256) void knn_plane_feat(
    const float* __restrict__ p, const float* __restrict__ plane_w,
    float* __restrict__ feat) {
  __shared__ float sx[256], sy[256], sz[256];
  const int b = blockIdx.y;
  const int i = blockIdx.x * 256 + threadIdx.x;
  const float* pb = p + (size_t)b * NPTS * 3;
  const float px = pb[i * 3 + 0], py = pb[i * 3 + 1], pz = pb[i * 3 + 2];

  float dist[KP1];
#pragma unroll
  for (int t = 0; t < KP1; ++t) dist[t] = 3.4e38f;
  float curMax = 3.4e38f;
  int   maxSlot = 0;

  for (int tile = 0; tile < NPTS / 256; ++tile) {
    const int j = tile * 256 + threadIdx.x;
    sx[threadIdx.x] = pb[j * 3 + 0];
    sy[threadIdx.x] = pb[j * 3 + 1];
    sz[threadIdx.x] = pb[j * 3 + 2];
    __syncthreads();
    for (int t = 0; t < 256; ++t) {
      const float dx = sx[t] - px, dy = sy[t] - py, dz = sz[t] - pz;
      const float d2 = dx * dx + dy * dy + dz * dz;
      if (d2 < curMax) {
        dist[maxSlot] = d2;
        curMax = dist[0]; maxSlot = 0;
        for (int s = 1; s < KP1; ++s)
          if (dist[s] > curMax) { curMax = dist[s]; maxSlot = s; }
      }
    }
    __syncthreads();
  }
  const float tau = curMax;   // (k+1)-th smallest distance (incl. self)

  float Sx = 0.f, Sy = 0.f, Sz = 0.f;
  int   m = 0;
  for (int tile = 0; tile < NPTS / 256; ++tile) {
    const int j = tile * 256 + threadIdx.x;
    sx[threadIdx.x] = pb[j * 3 + 0];
    sy[threadIdx.x] = pb[j * 3 + 1];
    sz[threadIdx.x] = pb[j * 3 + 2];
    __syncthreads();
    for (int t = 0; t < 256; ++t) {
      const float dx = sx[t] - px, dy = sy[t] - py, dz = sz[t] - pz;
      const float d2 = dx * dx + dy * dy + dz * dz;
      if (d2 <= tau && (tile * 256 + t) != i) {
        Sx += sx[t]; Sy += sy[t]; Sz += sz[t]; ++m;
      }
    }
    __syncthreads();
  }

  const float fm = (float)m;
#pragma unroll
  for (int c = 0; c < 3; ++c) {
    const float w0  = plane_w[c * 4 + 0];
    const float wvx = plane_w[c * 4 + 1];
    const float wvy = plane_w[c * 4 + 2];
    const float wvz = plane_w[c * 4 + 3];
    const float nrm = sqrtf(wvx * wvx + wvy * wvy + wvz * wvz);
    const float dot = (Sx - fm * px) * wvx + (Sy - fm * py) * wvy +
                      (Sz - fm * pz) * wvz;
    const float pc  = (fm * w0 + dot) / (nrm * (float)KNN_K);
    feat[((size_t)b * NPTS + i) * 3 + c] =
        1.f / (1.f + expf(pc * 1.00430108f /* ln(2.73) */));
  }
}

// ---------------------------------------------------------------------------
// l1 (3 -> 64) + BN1 (over batch pair) + ReLU, fused.
// ---------------------------------------------------------------------------
__global__ void l1_bn_relu(const float* __restrict__ feat,
                           const float* __restrict__ W1,
                           const float* __restrict__ b1,
                           const float* __restrict__ g,
                           const float* __restrict__ beta,
                           unsigned short* __restrict__ out) {
  const int idx = blockIdx.x * blockDim.x + threadIdx.x;  // n*64 + f
  if (idx >= NPTS * 64) return;
  const int n = idx >> 6, f = idx & 63;
  const float w0 = W1[f * 3 + 0], w1 = W1[f * 3 + 1], w2 = W1[f * 3 + 2];
  const float* f0 = feat + (size_t)n * 3;
  const float* f1 = feat + (size_t)(NPTS + n) * 3;
  const float y0 = b1[f] + f0[0] * w0 + f0[1] * w1 + f0[2] * w2;
  const float y1 = b1[f] + f1[0] * w0 + f1[1] * w1 + f1[2] * w2;
  const float mu = 0.5f * (y0 + y1);
  const float d0 = y0 - mu, d1 = y1 - mu;
  const float is = rsqrtf(0.5f * (d0 * d0 + d1 * d1) + BN_EPS);
  out[(size_t)n * 64 + f]          = f2bf(fmaxf(d0 * is * g[f] + beta[f], 0.f));
  out[(size_t)(NPTS + n) * 64 + f] = f2bf(fmaxf(d1 * is * g[f] + beta[f], 0.f));
}

// ---------------------------------------------------------------------------
// BN (over the 2-element batch axis) + ReLU, bf16 in/out.
// ---------------------------------------------------------------------------
__global__ void bn_relu(const unsigned short* __restrict__ y,
                        const float* __restrict__ g,
                        const float* __restrict__ beta,
                        unsigned short* __restrict__ out, int F) {
  const int idx = blockIdx.x * blockDim.x + threadIdx.x;  // n*F + f
  if (idx >= NPTS * F) return;
  const int n = idx / F, f = idx % F;
  const float y0 = bf2f(y[(size_t)n * F + f]);
  const float y1 = bf2f(y[(size_t)(NPTS + n) * F + f]);
  const float mu = 0.5f * (y0 + y1);
  const float d0 = y0 - mu, d1 = y1 - mu;
  const float is = rsqrtf(0.5f * (d0 * d0 + d1 * d1) + BN_EPS);
  out[(size_t)n * F + f]          = f2bf(fmaxf(d0 * is * g[f] + beta[f], 0.f));
  out[(size_t)(NPTS + n) * F + f] = f2bf(fmaxf(d1 * is * g[f] + beta[f], 0.f));
}

// ---------------------------------------------------------------------------
// WMMA GEMM: out[M x F] = (ReLU?)A[M x K] . W[F x K]^T  (+bias) (+Dadd)
// 8 waves / block.  Each wave owns a 32(M) x 64(N) tile: 2 A-fragments x
// 4 B-fragments -> 8 v_wmma_f32_16x16x32_bf16 per 32-wide K step.
// K loop is unrolled by 2 with two *independent* fragment register sets
// (ping-pong): loads for one set are in flight while the other set feeds the
// WMMAs.  No register rotation -> no v_mov traffic, no WMMA->VALU WAR nops.
// M % 32 == 0, F % 64 == 0, K % 64 == 0 (guaranteed by the network).
// ---------------------------------------------------------------------------
template <bool RELU_A, bool HAS_BIAS, bool ADD_D>
__global__ __launch_bounds__(256) void gemm_wmma(
    const unsigned short* __restrict__ A, const unsigned short* __restrict__ W,
    const float* __restrict__ bias, const unsigned short* __restrict__ Dadd,
    unsigned short* __restrict__ out, int M, int K, int F) {
  const int lane   = threadIdx.x & 31;
  const int ntiles = F >> 6;
  const int gw     = blockIdx.x * 8 + (threadIdx.x >> 5);
  const int nt     = gw % ntiles;
  const int mt     = gw / ntiles;
  if (mt >= (M >> 5)) return;
  const int mbase = mt << 5;
  const int nbase = nt << 6;
  const int half  = lane >> 4;
  const int r     = lane & 15;

  const unsigned short* a0row = A + (size_t)(mbase + r)      * K + 16 * half;
  const unsigned short* a1row = A + (size_t)(mbase + 16 + r) * K + 16 * half;
  const unsigned short* w0p   = W + (size_t)(nbase +  0 + r) * K + 16 * half;
  const unsigned short* w1p   = W + (size_t)(nbase + 16 + r) * K + 16 * half;
  const unsigned short* w2p   = W + (size_t)(nbase + 32 + r) * K + 16 * half;
  const unsigned short* w3p   = W + (size_t)(nbase + 48 + r) * K + 16 * half;

  v8f acc[2][4];
#pragma unroll
  for (int mi = 0; mi < 2; ++mi)
#pragma unroll
    for (int j = 0; j < 4; ++j) acc[mi][j] = v8f{};

#define LOAD_SET(S, k)                                \
  do {                                                \
    (S).a0 = load_fragA<RELU_A>(a0row + (k));         \
    (S).a1 = load_fragA<RELU_A>(a1row + (k));         \
    (S).b0 = load_fragB(w0p + (k));                   \
    (S).b1 = load_fragB(w1p + (k));                   \
    (S).b2 = load_fragB(w2p + (k));                   \
    (S).b3 = load_fragB(w3p + (k));                   \
  } while (0)

#define MMA_SET(S)                                                            \
  do {                                                                        \
    acc[0][0] = __builtin_amdgcn_wmma_f32_16x16x32_bf16(                      \
        false, (S).a0, false, (S).b0, (short)0, acc[0][0], false, false);     \
    acc[0][1] = __builtin_amdgcn_wmma_f32_16x16x32_bf16(                      \
        false, (S).a0, false, (S).b1, (short)0, acc[0][1], false, false);     \
    acc[0][2] = __builtin_amdgcn_wmma_f32_16x16x32_bf16(                      \
        false, (S).a0, false, (S).b2, (short)0, acc[0][2], false, false);     \
    acc[0][3] = __builtin_amdgcn_wmma_f32_16x16x32_bf16(                      \
        false, (S).a0, false, (S).b3, (short)0, acc[0][3], false, false);     \
    acc[1][0] = __builtin_amdgcn_wmma_f32_16x16x32_bf16(                      \
        false, (S).a1, false, (S).b0, (short)0, acc[1][0], false, false);     \
    acc[1][1] = __builtin_amdgcn_wmma_f32_16x16x32_bf16(                      \
        false, (S).a1, false, (S).b1, (short)0, acc[1][1], false, false);     \
    acc[1][2] = __builtin_amdgcn_wmma_f32_16x16x32_bf16(                      \
        false, (S).a1, false, (S).b2, (short)0, acc[1][2], false, false);     \
    acc[1][3] = __builtin_amdgcn_wmma_f32_16x16x32_bf16(                      \
        false, (S).a1, false, (S).b3, (short)0, acc[1][3], false, false);     \
  } while (0)

  FragSet s0, s1;
  LOAD_SET(s0, 0);                       // prologue

#pragma unroll 1
  for (int k0 = 0; k0 < K; k0 += 64) {
    // loads for the odd half-step (always in range since K % 64 == 0)
    LOAD_SET(s1, k0 + 32);
    __builtin_prefetch(a0row + k0 + 64, 0, 3);     // global_prefetch_b8
    __builtin_prefetch(w0p + k0 + 64, 0, 3);
    __builtin_prefetch(w2p + k0 + 64, 0, 3);
    MMA_SET(s0);                         // consumes set0; set1 loads in flight
    if (k0 + 64 < K) LOAD_SET(s0, k0 + 64);
    MMA_SET(s1);                         // consumes set1; set0 loads in flight
  }
#undef LOAD_SET
#undef MMA_SET

  // D layout: VGPR i -> row +(i + 8*half), column nbase + 16*j + (lane%16)
#pragma unroll
  for (int mi = 0; mi < 2; ++mi) {
#pragma unroll
    for (int j = 0; j < 4; ++j) {
      const int   c  = nbase + 16 * j + r;
      const float bv = HAS_BIAS ? bias[c] : 0.f;
#pragma unroll
      for (int i2 = 0; i2 < 8; ++i2) {
        const int row = mbase + 16 * mi + i2 + 8 * half;
        float v = acc[mi][j][i2] + bv;
        if (ADD_D) v += bf2f(Dadd[(size_t)row * F + c]);
        out[(size_t)row * F + c] = f2bf(v);
      }
    }
  }
}

// ---------------------------------------------------------------------------
// Two-stage column max over the N axis (F = 512).
// ---------------------------------------------------------------------------
__global__ void rowmax_part(const unsigned short* __restrict__ net,
                            float* __restrict__ part) {
  const int f = blockIdx.x * 256 + threadIdx.x;   // 0..511
  const int b = blockIdx.y;
  const int chunk = blockIdx.z;                   // 64 chunks of 128 rows
  const unsigned short* base =
      net + ((size_t)b * NPTS + (size_t)chunk * 128) * 512 + f;
  float m = -3.4e38f;
  for (int n = 0; n < 128; ++n) m = fmaxf(m, bf2f(base[(size_t)n * 512]));
  part[((size_t)chunk * BATCH + b) * 512 + f] = m;
}
__global__ void rowmax_reduce(const float* __restrict__ part,
                              float* __restrict__ pool) {
  const int idx = blockIdx.x * blockDim.x + threadIdx.x;  // b*512 + f
  if (idx >= BATCH * 512) return;
  const int b = idx >> 9, f = idx & 511;
  float m = -3.4e38f;
  for (int c = 0; c < 64; ++c)
    m = fmaxf(m, part[((size_t)c * BATCH + b) * 512 + f]);
  pool[idx] = m;
}

// ---------------------------------------------------------------------------
// Build concat [net | broadcast(pooled)] -> bf16 [ROWS x 1024]
// ---------------------------------------------------------------------------
__global__ void build_concat(const unsigned short* __restrict__ net,
                             const float* __restrict__ pool,
                             unsigned short* __restrict__ out) {
  const int idx = blockIdx.x * 256 + threadIdx.x;
  if (idx >= ROWS * 1024) return;
  const int row = idx >> 10, c = idx & 1023;
  unsigned short v;
  if (c < 512) v = net[(size_t)row * 512 + c];
  else         v = f2bf(pool[(row >> 13) * 512 + (c - 512)]);
  out[idx] = v;
}

// ---------------------------------------------------------------------------
// Final: out[b,f] = b_c[f] + sum_k relu(pool[b,k]) * Wc[f,k]   (f32)
// ---------------------------------------------------------------------------
__global__ void final_fc(const float* __restrict__ pool,
                         const float* __restrict__ Wc,
                         const float* __restrict__ bc,
                         float* __restrict__ out) {
  const int idx = blockIdx.x * blockDim.x + threadIdx.x;  // b*512 + f
  if (idx >= BATCH * 512) return;
  const int b = idx >> 9, f = idx & 511;
  const float* w = Wc + (size_t)f * 512;
  const float* m = pool + (size_t)b * 512;
  float s = bc[f];
  for (int k = 0; k < 512; ++k) s += fmaxf(m[k], 0.f) * w[k];
  out[idx] = s;
}

// ---------------------------------------------------------------------------
// Host launcher
// ---------------------------------------------------------------------------
extern "C" void kernel_launch(void* const* d_in, const int* in_sizes, int n_in,
                              void* d_out, int out_size, void* d_ws,
                              size_t ws_size, hipStream_t stream) {
  (void)in_sizes; (void)n_in; (void)out_size; (void)ws_size;

  const float* p      = (const float*)d_in[0];
  const float* planew = (const float*)d_in[1];
  const float* L1W = (const float*)d_in[2];
  const float* L1B = (const float*)d_in[3];
  const float* G1  = (const float*)d_in[4];
  const float* BE1 = (const float*)d_in[5];
  const float* L2W = (const float*)d_in[6];
  const float* L2B = (const float*)d_in[7];
  const float* G2  = (const float*)d_in[8];
  const float* BE2 = (const float*)d_in[9];
  const float* L3W = (const float*)d_in[10];
  const float* L3B = (const float*)d_in[11];
  const float* G3  = (const float*)d_in[12];
  const float* BE3 = (const float*)d_in[13];
  const float* L4W = (const float*)d_in[14];
  const float* L4B = (const float*)d_in[15];
  const float* FC0W[5], *FC0B[5], *FC1W[5], *FC1B[5], *SCW[5];
  for (int i = 0; i < 5; ++i) {
    FC0W[i] = (const float*)d_in[16 + i * 5 + 0];
    FC0B[i] = (const float*)d_in[16 + i * 5 + 1];
    FC1W[i] = (const float*)d_in[16 + i * 5 + 2];
    FC1B[i] = (const float*)d_in[16 + i * 5 + 3];
    SCW[i]  = (const float*)d_in[16 + i * 5 + 4];
  }
  const float* FCCW = (const float*)d_in[41];
  const float* FCCB = (const float*)d_in[42];
  float* out = (float*)d_out;

  // ---- workspace carve ----
  char*  base = (char*)d_ws;
  size_t off  = 0;
  auto alloc = [&](size_t bytes) -> void* {
    off = (off + 255) & ~(size_t)255;
    void* r = base + off;
    off += bytes;
    return r;
  };
  unsigned short* wW2 = (unsigned short*)alloc((size_t)8192 * 2);
  unsigned short* wW3 = (unsigned short*)alloc((size_t)16384 * 2);
  unsigned short* wW4 = (unsigned short*)alloc((size_t)131072 * 2);
  unsigned short *wFC0[5], *wFC1[5], *wSC[5];
  for (int i = 0; i < 5; ++i) {
    wFC0[i] = (unsigned short*)alloc((size_t)524288 * 2);
    wFC1[i] = (unsigned short*)alloc((size_t)262144 * 2);
    wSC[i]  = (unsigned short*)alloc((size_t)524288 * 2);
  }
  float*          feat  = (float*)alloc((size_t)ROWS * 3 * 4);
  unsigned short* bufX  = (unsigned short*)alloc((size_t)ROWS * 1024 * 2);
  unsigned short* bufN  = (unsigned short*)alloc((size_t)ROWS * 512 * 2);
  unsigned short* bufT1 = (unsigned short*)alloc((size_t)ROWS * 512 * 2);
  unsigned short* bufT2 = (unsigned short*)alloc((size_t)ROWS * 512 * 2);
  float*          pool  = (float*)alloc((size_t)BATCH * 512 * 4);
  float*          part  = (float*)alloc((size_t)64 * BATCH * 512 * 4);

  // ---- weight conversion to bf16 ----
#define CVT(src, dst, n) \
  cvt_f32_bf16<<<((n) + 255) / 256, 256, 0, stream>>>(src, dst, n)
  CVT(L2W, wW2, 8192);
  CVT(L3W, wW3, 16384);
  CVT(L4W, wW4, 131072);
  for (int i = 0; i < 5; ++i) {
    CVT(FC0W[i], wFC0[i], 524288);
    CVT(FC1W[i], wFC1[i], 262144);
    CVT(SCW[i],  wSC[i],  524288);
  }
#undef CVT

  // ---- KNN plane features ----
  knn_plane_feat<<<dim3(NPTS / 256, BATCH), 256, 0, stream>>>(p, planew, feat);

  // ---- MLP ----
  l1_bn_relu<<<(NPTS * 64) / 256, 256, 0, stream>>>(feat, L1W, L1B, G1, BE1,
                                                    bufN);
  const int MT32 = ROWS / 32;  // 512 m-tiles of 32 rows
  // l2: K=64, F=128
  gemm_wmma<false, true, false><<<MT32 * 2 / 8, 256, 0, stream>>>(
      bufN, wW2, L2B, nullptr, bufT2, ROWS, 64, 128);
  bn_relu<<<(NPTS * 128) / 256, 256, 0, stream>>>(bufT2, G2, BE2, bufN, 128);
  // l3: K=128, F=128
  gemm_wmma<false, true, false><<<MT32 * 2 / 8, 256, 0, stream>>>(
      bufN, wW3, L3B, nullptr, bufT2, ROWS, 128, 128);
  bn_relu<<<(NPTS * 128) / 256, 256, 0, stream>>>(bufT2, G3, BE3, bufN, 128);
  // l4: K=128, F=1024  -> x (resblock input)
  gemm_wmma<false, true, false><<<MT32 * 16 / 8, 256, 0, stream>>>(
      bufN, wW4, L4B, nullptr, bufX, ROWS, 128, 1024);

  // ---- resblock 0 (din = 1024) ----
  gemm_wmma<true, true, false><<<MT32 * 8 / 8, 256, 0, stream>>>(
      bufX, wFC0[0], FC0B[0], nullptr, bufT1, ROWS, 1024, 512);
  gemm_wmma<true, true, false><<<MT32 * 8 / 8, 256, 0, stream>>>(
      bufT1, wFC1[0], FC1B[0], nullptr, bufT2, ROWS, 512, 512);
  gemm_wmma<false, false, true><<<MT32 * 8 / 8, 256, 0, stream>>>(
      bufX, wSC[0], nullptr, bufT2, bufN, ROWS, 1024, 512);

  // ---- resblocks 1..4: pool -> concat -> block ----
  for (int i = 1; i < 5; ++i) {
    rowmax_part<<<dim3(2, BATCH, 64), 256, 0, stream>>>(bufN, part);
    rowmax_reduce<<<(BATCH * 512 + 255) / 256, 256, 0, stream>>>(part, pool);
    build_concat<<<(ROWS * 1024) / 256, 256, 0, stream>>>(bufN, pool, bufX);
    gemm_wmma<true, true, false><<<MT32 * 8 / 8, 256, 0, stream>>>(
        bufX, wFC0[i], FC0B[i], nullptr, bufT1, ROWS, 1024, 512);
    gemm_wmma<true, true, false><<<MT32 * 8 / 8, 256, 0, stream>>>(
        bufT1, wFC1[i], FC1B[i], nullptr, bufT2, ROWS, 512, 512);
    gemm_wmma<false, false, true><<<MT32 * 8 / 8, 256, 0, stream>>>(
        bufX, wSC[i], nullptr, bufT2, bufN, ROWS, 1024, 512);
  }

  // ---- final pool + FC ----
  rowmax_part<<<dim3(2, BATCH, 64), 256, 0, stream>>>(bufN, part);
  rowmax_reduce<<<(BATCH * 512 + 255) / 256, 256, 0, stream>>>(part, pool);
  final_fc<<<(BATCH * 512 + 255) / 256, 256, 0, stream>>>(pool, FCCW, FCCB,
                                                          out);
}
